// SchNet_1855425871946
// MI455X (gfx1250) — matrix operand order repaired and checked by reading, not compiled, not run on hardware
//
#include <hip/hip_runtime.h>

typedef __attribute__((ext_vector_type(16))) _Float16 v16h;
typedef __attribute__((ext_vector_type(8)))  float    v8f;

#define TPB 256
constexpr int   kH   = 128;   // hidden channels
constexpr int   kNF  = 128;   // num filters
constexpr int   kG   = 50;    // num gaussians
constexpr int   kGp  = 64;    // padded gaussians (multiple of 32 for WMMA K)
constexpr int   kL   = 6;     // interactions
constexpr float kCutoff = 10.0f;
constexpr float kLog2   = 0.69314718055994530942f;

// Shifted softplus, branch-free fast form:
//   softplus(x) = max(x,0) + log(1 + exp(-|x|))
// __expf/__logf lower to single v_exp_f32 / v_log_f32 (TRANS ops that
// co-execute with WMMA), and EXEC stays all-ones (required around WMMA).
__device__ __forceinline__ float ssp(float x) {
  float t = __expf(-fabsf(x));
  return fmaxf(x, 0.0f) + __logf(1.0f + t) - kLog2;
}

__device__ __forceinline__ v8f wmma16(v16h a, v16h b, v8f c) {
  // D(f32 16x16) = A(f16 16x32) * B(f16 32x16) + C
  return __builtin_amdgcn_wmma_f32_16x16x32_f16(false, a, false, b, (short)0, c,
                                                false, false);
}

// A fragment (16x32 f16) from LDS row-major tile, row m = lane&15.
// ISA layout: VGPR v holds K = 16*(v/4) + 8*half + 2*(v%4), +1.
__device__ __forceinline__ v16h frag_a(const _Float16* s, int m, int kk, int ldS,
                                       int half) {
  v16h a;
#pragma unroll
  for (int v = 0; v < 8; ++v) {
    int k = kk + 16 * (v >> 2) + 8 * half + 2 * (v & 3);
    a[2 * v]     = s[m * ldS + k];
    a[2 * v + 1] = s[m * ldS + k + 1];
  }
  return a;
}

// B fragment (32x16 f16) from global weight matrix [K x 128] row-major, column col.
// ISA layout: VGPR v holds K = 16*(v/4) + 8*half + 2*(v%4), +1 (matches frag_a).
__device__ __forceinline__ v16h frag_b(const _Float16* w, int kk, int col, int half) {
  v16h b;
#pragma unroll
  for (int v = 0; v < 8; ++v) {
    int k = kk + 16 * (v >> 2) + 8 * half + 2 * (v & 3);
    b[2 * v]     = w[k * kNF + col];
    b[2 * v + 1] = w[(k + 1) * kNF + col];
  }
  return b;
}

// ---------------- one-time prep kernels ----------------

// Convert all per-layer weights to f16; zero-pad mlp_W1 from [50,128] to [64,128].
__global__ void k_convw(const float* __restrict__ mW1, const float* __restrict__ mW2,
                        const float* __restrict__ cl1, const float* __restrict__ cl2,
                        const float* __restrict__ lw,
                        _Float16* __restrict__ W1h, _Float16* __restrict__ W2h,
                        _Float16* __restrict__ L1h, _Float16* __restrict__ L2h,
                        _Float16* __restrict__ LWh) {
  int idx = blockIdx.x * blockDim.x + threadIdx.x;
  const int t1 = kL * kGp * kNF;
  const int t2 = kL * kNF * kNF;
  if (idx < t1) {
    int l = idx / (kGp * kNF);
    int rem = idx % (kGp * kNF);
    int g = rem / kNF, n = rem % kNF;
    W1h[idx] = (g < kG) ? (_Float16)mW1[(l * kG + g) * kNF + n] : (_Float16)0.0f;
    return;
  }
  int j = idx - t1;
  if (j >= 4 * t2) return;
  int region = j / t2, r = j % t2;
  const float* src = (region == 0) ? mW2 : (region == 1) ? cl1
                   : (region == 2) ? cl2 : lw;
  _Float16* dst = (region == 0) ? W2h : (region == 1) ? L1h
                : (region == 2) ? L2h : LWh;
  dst[r] = (_Float16)src[r];
}

// Gaussian smearing (f16, padded to 64) + cosine cutoff for the E unique edges.
__global__ void k_pre(const float* __restrict__ ew, const float* __restrict__ off,
                      _Float16* __restrict__ ea16, float* __restrict__ ccut, int E) {
  int idx = blockIdx.x * blockDim.x + threadIdx.x;
  if (idx >= E * kGp) return;
  int e = idx >> 6, k = idx & 63;
  float w = ew[e];
  float delta = off[1] - off[0];
  float coeff = -0.5f / (delta * delta);
  float v = 0.0f;
  if (k < kG) {
    float d = w - off[k];
    v = expf(coeff * d * d);
  }
  ea16[idx] = (_Float16)v;
  if (k == 0)
    ccut[e] = 0.5f * (cosf(w * 3.14159265358979323846f / kCutoff) + 1.0f);
}

// Embedding: DIN==1 -> h[n,:] = sites[n]*W_emb + b_emb
__global__ void k_embed(const float* __restrict__ sites, const float* __restrict__ Wemb,
                        const float* __restrict__ bemb, float* __restrict__ h,
                        int total) {
  int idx = blockIdx.x * blockDim.x + threadIdx.x;
  if (idx >= total) return;
  int c = idx & (kH - 1);
  h[idx] = sites[idx >> 7] * Wemb[c] + bemb[c];
}

// ---------------- per-layer kernels ----------------

// Fused edge-filter MLP: Wf = (ssp(ea @ W1 + b1) @ W2 + b2) * C.
// One block = 16 edges; 8 waves, wave w owns output columns [16w,16w+16).
__global__ __launch_bounds__(TPB) void k_filter(
    const _Float16* __restrict__ ea16, const float* __restrict__ ccut,
    const _Float16* __restrict__ W1, const float* __restrict__ b1,
    const _Float16* __restrict__ W2, const float* __restrict__ b2,
    float* __restrict__ Wf) {
  __shared__ _Float16 sA[16 * kGp];
  __shared__ _Float16 sT[16 * kNF];
  const int tid = threadIdx.x;
  const int e0 = blockIdx.x * 16;

  // stage 16x64 f16 edge attributes (2048 B) via 32-bit copies
  const unsigned int* g = (const unsigned int*)(ea16 + (size_t)e0 * kGp);
  unsigned int* sAi = (unsigned int*)sA;
  sAi[tid]       = g[tid];
  sAi[tid + TPB] = g[tid + TPB];
  __syncthreads();

  const int lane = tid & 31, wave = tid >> 5;
  const int half = lane >> 4, m = lane & 15;
  const int col = wave * 16 + m;

  v8f acc = {};
#pragma unroll
  for (int kk = 0; kk < kGp; kk += 32)
    acc = wmma16(frag_a(sA, m, kk, kGp, half), frag_b(W1, kk, col, half), acc);

  const float bia = b1[col];
#pragma unroll
  for (int r = 0; r < 8; ++r) {
    int row = 8 * half + r;
    sT[row * kNF + col] = (_Float16)ssp(acc[r] + bia);
  }
  __syncthreads();

  v8f acc2 = {};
#pragma unroll
  for (int kk = 0; kk < kNF; kk += 32)
    acc2 = wmma16(frag_a(sT, m, kk, kNF, half), frag_b(W2, kk, col, half), acc2);

  const float bia2 = b2[col];
#pragma unroll
  for (int r = 0; r < 8; ++r) {
    int row = 8 * half + r;
    Wf[(size_t)(e0 + row) * kNF + col] = (acc2[r] + bia2) * ccut[e0 + row];
  }
}

// xf = h @ conv_lin1_W   (no bias). One block = 16 node rows.
__global__ __launch_bounds__(TPB) void k_lin1(const float* __restrict__ h,
                                              const _Float16* __restrict__ W,
                                              float* __restrict__ xf) {
  __shared__ _Float16 sH[16 * kH];
  const int tid = threadIdx.x;
  const int r0 = blockIdx.x * 16;
  const float* hg = h + (size_t)r0 * kH;
#pragma unroll
  for (int i = 0; i < 8; ++i) {
    int idx = tid + TPB * i;
    sH[idx] = (_Float16)hg[idx];
  }
  __syncthreads();
  const int lane = tid & 31, wave = tid >> 5;
  const int half = lane >> 4, m = lane & 15;
  const int col = wave * 16 + m;
  v8f acc = {};
#pragma unroll
  for (int kk = 0; kk < kH; kk += 32)
    acc = wmma16(frag_a(sH, m, kk, kH, half), frag_b(W, kk, col, half), acc);
#pragma unroll
  for (int r = 0; r < 8; ++r) {
    int row = 8 * half + r;
    xf[(size_t)(r0 + row) * kNF + col] = acc[r];
  }
}

// Message pass: agg[dst] += xf[src] * Wf[e % E]  (Wf shared across batch replicas).
__global__ void k_msg(const long long* __restrict__ ei, const float* __restrict__ xf,
                      const float* __restrict__ Wf, float* __restrict__ agg, int E,
                      int at, int total) {
  int idx = blockIdx.x * blockDim.x + threadIdx.x;
  if (idx >= total) return;
  int eb = idx >> 5;         // batched edge
  int q  = idx & 31;         // float4 slot within 128 channels
  int e  = eb % E;
  int b  = eb / E;
  int src = (int)ei[2 * e]     + b * at;
  int dst = (int)ei[2 * e + 1] + b * at;
  float4 wv = ((const float4*)(Wf + (size_t)e * kNF))[q];
  float4 xv = ((const float4*)(xf + (size_t)src * kNF))[q];
  float* ap = agg + (size_t)dst * kNF + q * 4;
  atomicAdd(ap + 0, wv.x * xv.x);
  atomicAdd(ap + 1, wv.y * xv.y);
  atomicAdd(ap + 2, wv.z * xv.z);
  atomicAdd(ap + 3, wv.w * xv.w);
}

// Fused node update: h += (ssp(agg @ W2c + b2c)) @ Wl + bl
__global__ __launch_bounds__(TPB) void k_nodeup(
    const float* __restrict__ agg, const _Float16* __restrict__ W2c,
    const float* __restrict__ b2c, const _Float16* __restrict__ Wl,
    const float* __restrict__ bl, float* __restrict__ h) {
  __shared__ _Float16 sA[16 * kH];
  __shared__ _Float16 sT[16 * kH];
  const int tid = threadIdx.x;
  const int r0 = blockIdx.x * 16;
  const float* ag = agg + (size_t)r0 * kH;
#pragma unroll
  for (int i = 0; i < 8; ++i) {
    int idx = tid + TPB * i;
    sA[idx] = (_Float16)ag[idx];
  }
  __syncthreads();
  const int lane = tid & 31, wave = tid >> 5;
  const int half = lane >> 4, m = lane & 15;
  const int col = wave * 16 + m;

  v8f acc = {};
#pragma unroll
  for (int kk = 0; kk < kH; kk += 32)
    acc = wmma16(frag_a(sA, m, kk, kH, half), frag_b(W2c, kk, col, half), acc);
  const float bia = b2c[col];
#pragma unroll
  for (int r = 0; r < 8; ++r) {
    int row = 8 * half + r;
    sT[row * kH + col] = (_Float16)ssp(acc[r] + bia);
  }
  __syncthreads();

  v8f acc2 = {};
#pragma unroll
  for (int kk = 0; kk < kH; kk += 32)
    acc2 = wmma16(frag_a(sT, m, kk, kH, half), frag_b(Wl, kk, col, half), acc2);
  const float bia2 = bl[col];
#pragma unroll
  for (int r = 0; r < 8; ++r) {
    int row = 8 * half + r;
    h[(size_t)(r0 + row) * kH + col] += acc2[r] + bia2;
  }
}

// ---------------- readout ----------------
// act is a no-op in the reference, so fold: w = l1_W @ l2_W,  cterm = l1_b.l2_W + l2_b
__global__ void k_read1(const float* __restrict__ l1W, const float* __restrict__ l1b,
                        const float* __restrict__ l2W, const float* __restrict__ l2b,
                        float* __restrict__ wv) {
  int c = threadIdx.x;
  float s = 0.0f;
  for (int j = 0; j < kH / 2; ++j) s += l1W[c * (kH / 2) + j] * l2W[j];
  wv[c] = s;
  if (c == 0) {
    float t = 0.0f;
    for (int j = 0; j < kH / 2; ++j) t += l1b[j] * l2W[j];
    wv[kH] = t + l2b[0];
  }
}

__global__ __launch_bounds__(TPB) void k_read2(const float* __restrict__ h,
                                               const float* __restrict__ wv,
                                               float* __restrict__ out, int at) {
  __shared__ float swv[kH + 1];
  __shared__ float red[TPB];
  int tid = threadIdx.x;
  if (tid < kH + 1) swv[tid] = wv[tid];
  __syncthreads();
  float acc = 0.0f;
  for (int a = tid; a < at; a += TPB) {
    const float* hr = h + ((size_t)blockIdx.x * at + a) * kH;
    float d = 0.0f;
#pragma unroll 4
    for (int c = 0; c < kH; ++c) d += hr[c] * swv[c];
    acc += d;
  }
  red[tid] = acc;
  __syncthreads();
  for (int s = TPB / 2; s > 0; s >>= 1) {
    if (tid < s) red[tid] += red[tid + s];
    __syncthreads();
  }
  if (tid == 0) out[blockIdx.x] = red[0] + (float)at * swv[kH];
}

// ---------------- driver ----------------

extern "C" void kernel_launch(void* const* d_in, const int* in_sizes, int n_in,
                              void* d_out, int out_size, void* d_ws, size_t ws_size,
                              hipStream_t stream) {
  (void)n_in; (void)ws_size;
  const float*     sites = (const float*)d_in[0];
  const long long* eidx  = (const long long*)d_in[1];   // int64 in reference
  const float*     ew    = (const float*)d_in[2];
  const float*     Wemb  = (const float*)d_in[3];
  const float*     bemb  = (const float*)d_in[4];
  const float*     offs  = (const float*)d_in[5];
  const float*     mW1   = (const float*)d_in[6];
  const float*     mb1   = (const float*)d_in[7];
  const float*     mW2   = (const float*)d_in[8];
  const float*     mb2   = (const float*)d_in[9];
  const float*     cl1W  = (const float*)d_in[10];
  const float*     cl2W  = (const float*)d_in[11];
  const float*     cl2b  = (const float*)d_in[12];
  const float*     linW  = (const float*)d_in[13];
  const float*     linb  = (const float*)d_in[14];
  const float*     l1W   = (const float*)d_in[15];
  const float*     l1b   = (const float*)d_in[16];
  const float*     l2W   = (const float*)d_in[17];
  const float*     l2b   = (const float*)d_in[18];
  float* out = (float*)d_out;

  const int N  = in_sizes[0];        // bs*at (DIN==1)
  const int E  = in_sizes[1] / 2;    // unique edges
  const int bs = out_size;
  const int at = N / bs;

  // carve workspace (256B aligned)
  char* p = (char*)d_ws;
  auto take = [&](size_t bytes) {
    char* q = p;
    p += (bytes + 255) & ~size_t(255);
    return q;
  };
  float*    h    = (float*)take((size_t)N * kH * 4);
  float*    xf   = (float*)take((size_t)N * kNF * 4);
  float*    agg  = (float*)take((size_t)N * kNF * 4);
  float*    Wf   = (float*)take((size_t)E * kNF * 4);
  _Float16* ea16 = (_Float16*)take((size_t)E * kGp * 2);
  float*    ccut = (float*)take((size_t)E * 4);
  _Float16* W1h  = (_Float16*)take((size_t)kL * kGp * kNF * 2);
  _Float16* W2h  = (_Float16*)take((size_t)kL * kNF * kNF * 2);
  _Float16* L1h  = (_Float16*)take((size_t)kL * kH * kNF * 2);
  _Float16* L2h  = (_Float16*)take((size_t)kL * kNF * kH * 2);
  _Float16* LWh  = (_Float16*)take((size_t)kL * kH * kH * 2);
  float*    wv   = (float*)take((kH + 1) * 4);

  {
    int total = kL * kGp * kNF + 4 * kL * kNF * kNF;
    k_convw<<<(total + TPB - 1) / TPB, TPB, 0, stream>>>(mW1, mW2, cl1W, cl2W, linW,
                                                         W1h, W2h, L1h, L2h, LWh);
  }
  k_pre<<<(E * kGp + TPB - 1) / TPB, TPB, 0, stream>>>(ew, offs, ea16, ccut, E);
  k_embed<<<(N * kH + TPB - 1) / TPB, TPB, 0, stream>>>(sites, Wemb, bemb, h, N * kH);

  const int msg_total = bs * E * 32;   // (batched edges) x (float4 slots)
  for (int l = 0; l < kL; ++l) {
    k_filter<<<E / 16, TPB, 0, stream>>>(ea16, ccut, W1h + (size_t)l * kGp * kNF,
                                         mb1 + l * kNF, W2h + (size_t)l * kNF * kNF,
                                         mb2 + l * kNF, Wf);
    k_lin1<<<N / 16, TPB, 0, stream>>>(h, L1h + (size_t)l * kH * kNF, xf);
    hipMemsetAsync(agg, 0, (size_t)N * kNF * 4, stream);
    k_msg<<<(msg_total + TPB - 1) / TPB, TPB, 0, stream>>>(eidx, xf, Wf, agg, E, at,
                                                           msg_total);
    k_nodeup<<<N / 16, TPB, 0, stream>>>(agg, L2h + (size_t)l * kNF * kH,
                                         cl2b + l * kH, LWh + (size_t)l * kH * kH,
                                         linb + l * kH, h);
  }
  k_read1<<<1, kH, 0, stream>>>(l1W, l1b, l2W, l2b, wv);
  k_read2<<<bs, TPB, 0, stream>>>(h, wv, out, at);
}
